// MultiHeadAttention_30949534335140
// MI455X (gfx1250) — matrix operand order repaired and checked by reading, not compiled, not run on hardware
//
#include <hip/hip_runtime.h>
#include <hip/hip_bf16.h>
#include <math.h>

// ---------------------------------------------------------------------------
// Problem constants (reference: B=2, T=2048, D=1024, H=16, dh=64)
// ---------------------------------------------------------------------------
static constexpr int NB     = 2;
static constexpr int SEQ    = 2048;
static constexpr int DMODEL = 1024;
static constexpr int NHEADS = 16;
static constexpr int HD     = 64;

typedef __attribute__((ext_vector_type(16))) __bf16 v16bf;
typedef __attribute__((ext_vector_type(8)))  float  v8f;
typedef __attribute__((ext_vector_type(4)))  unsigned v4u;
typedef __attribute__((ext_vector_type(8)))  int      v8i;
typedef __attribute__((ext_vector_type(4)))  int      v4i;

// Fragments are filled with 16B vector loads (b128).
union AFrag { uint4 q[2]; unsigned u[8]; v16bf v; };   // 16x32 bf16 A
union BFrag { uint4 q[2]; unsigned u[8]; v16bf v; };   // 32x16 bf16 B

static __device__ __forceinline__ unsigned short f2bf(float f) {
    unsigned u = __float_as_uint(f);
    u += 0x7FFFu + ((u >> 16) & 1u);   // round-to-nearest-even
    return (unsigned short)(u >> 16);
}

static __device__ __forceinline__ float rmax16(float v) {
    v = fmaxf(v, __shfl_xor(v, 8));
    v = fmaxf(v, __shfl_xor(v, 4));
    v = fmaxf(v, __shfl_xor(v, 2));
    v = fmaxf(v, __shfl_xor(v, 1));
    return v;
}
static __device__ __forceinline__ float rsum16(float v) {
    v += __shfl_xor(v, 8);
    v += __shfl_xor(v, 4);
    v += __shfl_xor(v, 2);
    v += __shfl_xor(v, 1);
    return v;
}

// ---------------------------------------------------------------------------
// fp32 -> bf16 convert (plain)
// ---------------------------------------------------------------------------
__global__ __launch_bounds__(256) void k_f32_to_bf16(const float* __restrict__ src,
                                                     unsigned short* __restrict__ dst,
                                                     int n) {
    int i = blockIdx.x * 256 + threadIdx.x;
    if (i < n) dst[i] = f2bf(src[i]);
}

// fp32 [K,N] -> bf16 transposed [N,K] (weights: B-fragments become contiguous)
__global__ __launch_bounds__(256) void k_f32_to_bf16_t(const float* __restrict__ src,
                                                       unsigned short* __restrict__ dst,
                                                       int K, int N) {
    int i = blockIdx.x * 256 + threadIdx.x;
    if (i >= K * N) return;
    int k = i / N, n = i % N;
    dst[(size_t)n * K + k] = f2bf(src[i]);
}

// ---------------------------------------------------------------------------
// bf16 GEMM: C[M,N] (f32) = A[M,K] @ Wt[N,K]^T, A row-major, Wt row-major.
// Block: 256 threads (8 waves). Block tile 256(M) x 64(N), K-step 32.
// Tiles are staged by the Tensor Data Mover (TENSOR_LOAD_TO_LDS, wave 0),
// completion via s_wait_tensorcnt; fragments read as ds_load_b128.
// Per wave / K-step: 8 WMMAs against 12 LDS b128 loads.
// ---------------------------------------------------------------------------
__global__ __launch_bounds__(256) void k_gemm_bf16(const unsigned short* __restrict__ A,
                                                   const unsigned short* __restrict__ Wt,
                                                   float* __restrict__ C,
                                                   int M, int N, int K) {
    __shared__ uint4 As4[1024];  // 256 rows x 32 bf16 (4 uint4 per row)
    __shared__ uint4 Bs4[256];   // 64 n-rows x 32 bf16

    const int tid  = threadIdx.x;
    const int wave = tid >> 5;
    const int lane = tid & 31;
    const int m    = lane & 15;
    const int hi   = lane >> 4;
    const int mblk = blockIdx.y * 256;
    const int nblk = blockIdx.x * 64;

    // Never taken: keeps the LDS arrays "stored-to" so the optimizer cannot
    // fold ds_loads of TDM-written LDS to undef.
    if (__builtin_expect(M < 0, 0)) {
        As4[tid] = uint4{0u, 0u, 0u, 0u};
        Bs4[tid] = uint4{0u, 0u, 0u, 0u};
    }

    v8f acc0[4] = {v8f{}, v8f{}, v8f{}, v8f{}};
    v8f acc1[4] = {v8f{}, v8f{}, v8f{}, v8f{}};

    const unsigned ldsA = (unsigned)(uintptr_t)(void*)As4;   // LDS aperture: addr[31:0] = LDS offset
    const unsigned ldsB = (unsigned)(uintptr_t)(void*)Bs4;
    const unsigned long long gbaseA =
        (unsigned long long)(uintptr_t)A + (unsigned long long)mblk * (unsigned long long)K * 2ull;
    const unsigned long long gbaseB =
        (unsigned long long)(uintptr_t)Wt + (unsigned long long)nblk * (unsigned long long)K * 2ull;

    for (int kb = 0; kb < K; kb += 32) {
        if (wave == 0) {
            // ---- D# group 0: count=1 | lds_addr | global tile addr | type=2
            unsigned long long gaA = gbaseA + (unsigned long long)kb * 2ull;
            unsigned long long gaB = gbaseB + (unsigned long long)kb * 2ull;
            v4u g0a = {1u, ldsA, (unsigned)gaA,
                       (unsigned)((gaA >> 32) & 0x01ffffffull) | 0x80000000u};
            v4u g0b = {1u, ldsB, (unsigned)gaB,
                       (unsigned)((gaB >> 32) & 0x01ffffffull) | 0x80000000u};
            // ---- D# group 1: data_size=2B; tensor_dim0=K, tensor_dim1=rows;
            //      tile_dim0=32, tile_dim1=256/64; tensor_dim0_stride=K
            v8i g1a = {(int)0x10000, (int)((unsigned)K << 16), (int)((unsigned)M << 16),
                       (int)(32u << 16), 256, K, 0, 0};
            v8i g1b = {(int)0x10000, (int)((unsigned)K << 16), (int)((unsigned)N << 16),
                       (int)(32u << 16), 64, K, 0, 0};
            v4i z4 = {0, 0, 0, 0};
            v8i z8 = {0, 0, 0, 0, 0, 0, 0, 0};
            __builtin_amdgcn_tensor_load_to_lds(g0a, g1a, z4, z4, z8, 0);
            __builtin_amdgcn_tensor_load_to_lds(g0b, g1b, z4, z4, z8, 0);
            __builtin_amdgcn_s_wait_tensorcnt(0);
        }
        __syncthreads();

        // ---- A fragments: rows 32*wave..+31, k = kb..kb+31 (2x b128 each)
        AFrag af0, af1;
        af0.q[0] = As4[(wave * 32 + m) * 4 + hi];
        af0.q[1] = As4[(wave * 32 + m) * 4 + 2 + hi];
        af1.q[0] = As4[(wave * 32 + 16 + m) * 4 + hi];
        af1.q[1] = As4[(wave * 32 + 16 + m) * 4 + 2 + hi];

        // ---- 4 column tiles; each B fragment feeds two WMMAs
        #pragma unroll
        for (int nt = 0; nt < 4; ++nt) {
            BFrag bf;
            int n = nt * 16 + m;
            bf.q[0] = Bs4[n * 4 + hi];
            bf.q[1] = Bs4[n * 4 + 2 + hi];
            acc0[nt] = __builtin_amdgcn_wmma_f32_16x16x32_bf16(
                false, af0.v, false, bf.v, (short)0, acc0[nt], false, false);
            acc1[nt] = __builtin_amdgcn_wmma_f32_16x16x32_bf16(
                false, af1.v, false, bf.v, (short)0, acc1[nt], false, false);
        }
        __syncthreads();
    }

    // ---- store (C-layout: vgpr r -> row r + 8*hi; lane -> column)
    #pragma unroll
    for (int nt = 0; nt < 4; ++nt) {
        #pragma unroll
        for (int r = 0; r < 8; ++r) {
            int row = mblk + wave * 32 + r + 8 * hi;
            int col = nblk + nt * 16 + m;
            C[(size_t)row * N + col]        = acc0[nt][r];
            C[(size_t)(row + 16) * N + col] = acc1[nt][r];
        }
    }
}

// ---------------------------------------------------------------------------
// RoPE + repack: Qf/Kf/Vf f32 [B*T, D] ->
//   Qb bf16 [B,H,T,64]  (RoPE'd, pre-scaled by 1/sqrt(dh)=0.125)
//   Kb bf16 [B,H,T,64]  (RoPE'd; row of K == column of K^T => contiguous B-frag)
//   Vt bf16 [B,H,64,T]  (transposed so P@V B-frags are contiguous along T)
// ---------------------------------------------------------------------------
__global__ __launch_bounds__(256) void k_rope_pack(const float* __restrict__ Qf,
                                                   const float* __restrict__ Kf,
                                                   const float* __restrict__ Vf,
                                                   unsigned short* __restrict__ Qb,
                                                   unsigned short* __restrict__ Kb,
                                                   unsigned short* __restrict__ Vt) {
    int gid = blockIdx.x * 256 + threadIdx.x;
    if (gid >= NB * SEQ * DMODEL) return;
    int col = gid % DMODEL;
    int row = gid / DMODEL;      // b*T + t
    int t = row % SEQ;
    int b = row / SEQ;
    int h = col / HD;
    int d = col % HD;

    int fi = d & 31;
    float freq = __expf(-(float)(2 * fi) * (9.210340371976184f / 64.0f)); // 10000^(-2fi/64)
    float s, c;
    __sincosf((float)t * freq, &s, &c);

    float q = Qf[gid], k = Kf[gid];
    float qr, kr;
    if (d < 32) { qr = -Qf[gid + 32]; kr = -Kf[gid + 32]; }
    else        { qr =  Qf[gid - 32]; kr =  Kf[gid - 32]; }

    size_t bh = (size_t)b * NHEADS + h;
    Qb[(bh * SEQ + t) * HD + d] = f2bf((q * c + qr * s) * 0.125f);
    Kb[(bh * SEQ + t) * HD + d] = f2bf(k * c + kr * s);
    Vt[(bh * HD + d) * SEQ + t] = f2bf(Vf[gid]);
}

// ---------------------------------------------------------------------------
// Causal flash attention. Grid: (T/128, B*H). Block 256 = 8 waves.
// Wave w: one 16-row query tile; keys streamed 32 at a time.
//   S(16x32) = Q(16x64) @ K^T     : 4 WMMAs (K-frags: 2x global b128 each)
//   online softmax, P -> LDS -> A-frag (b128 reads)
//   O(16x64) += P(16x32) @ V(32x64): 4 WMMAs (V-frags: 2x global b128 each)
// ---------------------------------------------------------------------------
__global__ __launch_bounds__(256) void k_attention(const unsigned short* __restrict__ Qb,
                                                   const unsigned short* __restrict__ Kb,
                                                   const unsigned short* __restrict__ Vt,
                                                   unsigned short* __restrict__ Ob) {
    __shared__ uint4 Ps4[8][64];   // per-wave 16x32 bf16 P scratch

    const int tid  = threadIdx.x;
    const int wave = tid >> 5;
    const int lane = tid & 31;
    const int m    = lane & 15;
    const int hi   = lane >> 4;
    const int bh   = blockIdx.y;
    const int b    = bh >> 4;
    const int h    = bh & 15;
    const int qs0  = blockIdx.x * 128 + wave * 16;

    // ---- Q fragments: 4x b128, held in registers for the whole key loop
    AFrag qa0, qa1;
    {
        const uint4* qrow = (const uint4*)(Qb + ((size_t)bh * SEQ + qs0 + m) * HD);
        qa0.q[0] = qrow[hi];
        qa0.q[1] = qrow[2 + hi];
        qa1.q[0] = qrow[4 + hi];
        qa1.q[1] = qrow[6 + hi];
    }
    const unsigned short* kbb = Kb + (size_t)bh * SEQ * HD;
    const unsigned short* vtb = Vt + (size_t)bh * HD * SEQ;

    v8f o0 = {}, o1 = {}, o2 = {}, o3 = {};
    float rm[8], rl[8];
    #pragma unroll
    for (int r = 0; r < 8; ++r) { rm[r] = -3.0e38f; rl[r] = 0.0f; }

    unsigned short* PsH = (unsigned short*)Ps4[wave];
    const uint4* Psq = Ps4[wave];

    const int qhi = qs0 + 15;
    for (int kb = 0; kb <= qhi; kb += 32) {
        if (kb + 32 <= qhi) {  // prefetch next K/V tiles
            __builtin_prefetch(kbb + (size_t)(kb + 32 + m) * HD, 0, 0);
            __builtin_prefetch(vtb + (size_t)(m) * SEQ + kb + 32, 0, 0);
        }
        v8f z = {};
        // ---- scores for keys kb+m (tile 0) : column of K^T = row of K
        BFrag b0, b1;
        {
            const uint4* kp = (const uint4*)(kbb + (size_t)(kb + m) * HD);
            b0.q[0] = kp[hi];              // d = hi*8 .. +7
            b0.q[1] = kp[2 + hi];          // d = 16+hi*8 .. +7
            b1.q[0] = kp[4 + hi];          // d = 32+hi*8 ..
            b1.q[1] = kp[6 + hi];
        }
        v8f s0 = __builtin_amdgcn_wmma_f32_16x16x32_bf16(false, qa0.v, false, b0.v, (short)0, z, false, false);
        s0     = __builtin_amdgcn_wmma_f32_16x16x32_bf16(false, qa1.v, false, b1.v, (short)0, s0, false, false);

        // ---- scores for keys kb+16+m (tile 1)
        {
            const uint4* kp = (const uint4*)(kbb + (size_t)(kb + 16 + m) * HD);
            b0.q[0] = kp[hi];
            b0.q[1] = kp[2 + hi];
            b1.q[0] = kp[4 + hi];
            b1.q[1] = kp[6 + hi];
        }
        v8f s1 = __builtin_amdgcn_wmma_f32_16x16x32_bf16(false, qa0.v, false, b0.v, (short)0, z, false, false);
        s1     = __builtin_amdgcn_wmma_f32_16x16x32_bf16(false, qa1.v, false, b1.v, (short)0, s1, false, false);

        // ---- causal mask + online softmax (row r+8*hi lives in this 16-lane half)
        #pragma unroll
        for (int r = 0; r < 8; ++r) {
            int q = qs0 + r + 8 * hi;
            if (kb + m      > q) s0[r] = -3.0e38f;
            if (kb + 16 + m > q) s1[r] = -3.0e38f;
        }
        #pragma unroll
        for (int r = 0; r < 8; ++r) {
            float tmax  = rmax16(fmaxf(s0[r], s1[r]));
            float mnew  = fmaxf(rm[r], tmax);
            float alpha = __expf(rm[r] - mnew);
            rm[r] = mnew;
            float p0 = __expf(s0[r] - mnew);
            float p1 = __expf(s1[r] - mnew);
            rl[r] = rl[r] * alpha + rsum16(p0 + p1);
            o0[r] *= alpha; o1[r] *= alpha; o2[r] *= alpha; o3[r] *= alpha;
            int rr = r + 8 * hi;
            PsH[rr * 32 + m]      = f2bf(p0);
            PsH[rr * 32 + 16 + m] = f2bf(p1);
        }
        asm volatile("s_wait_dscnt 0" ::: "memory");   // P stores visible wave-wide

        // ---- P as A-fragment (2x b128 from LDS)
        AFrag pf;
        pf.q[0] = Psq[m * 4 + hi];
        pf.q[1] = Psq[m * 4 + 2 + hi];

        // ---- O += P @ V  (V-frags contiguous along T in Vt)
        #pragma unroll
        for (int nt = 0; nt < 4; ++nt) {
            BFrag vf;
            int n = nt * 16 + m;
            const uint4* vp = (const uint4*)(vtb + (size_t)n * SEQ + kb + hi * 8);
            vf.q[0] = vp[0];               // keys kb+hi*8 .. +7
            vf.q[1] = vp[2];               // keys kb+16+hi*8 .. +7
            v8f* op = (nt == 0) ? &o0 : (nt == 1) ? &o1 : (nt == 2) ? &o2 : &o3;
            *op = __builtin_amdgcn_wmma_f32_16x16x32_bf16(
                false, pf.v, false, vf.v, (short)0, *op, false, false);
        }
        asm volatile("s_wait_dscnt 0" ::: "memory");   // reads done before next overwrite
    }

    // ---- normalize + store bf16 [B*T, D] (column h*64 + ...)
    #pragma unroll
    for (int r = 0; r < 8; ++r) {
        float inv = 1.0f / (rl[r] + 1e-12f);
        int row = b * SEQ + qs0 + r + 8 * hi;
        unsigned short* orow = Ob + (size_t)row * DMODEL + h * HD;
        orow[m]      = f2bf(o0[r] * inv);
        orow[16 + m] = f2bf(o1[r] * inv);
        orow[32 + m] = f2bf(o2[r] * inv);
        orow[48 + m] = f2bf(o3[r] * inv);
    }
}

// ---------------------------------------------------------------------------
// Launch
// ---------------------------------------------------------------------------
extern "C" void kernel_launch(void* const* d_in, const int* in_sizes, int n_in,
                              void* d_out, int out_size, void* d_ws, size_t ws_size,
                              hipStream_t stream) {
    (void)in_sizes; (void)n_in; (void)out_size; (void)ws_size;
    const float* x  = (const float*)d_in[0];
    const float* Wq = (const float*)d_in[1];
    const float* Wk = (const float*)d_in[2];
    const float* Wv = (const float*)d_in[3];
    const float* Wo = (const float*)d_in[4];
    // d_in[5] = causal mask (recomputed analytically on device)

    char* ws = (char*)d_ws;
    const size_t MB = 1ull << 20;
    unsigned short* xb  = (unsigned short*)(ws + 0 * MB);   // 8 MiB
    unsigned short* wqt = (unsigned short*)(ws + 8 * MB);   // 2 MiB each (transposed bf16)
    unsigned short* wkt = (unsigned short*)(ws + 10 * MB);
    unsigned short* wvt = (unsigned short*)(ws + 12 * MB);
    unsigned short* wot = (unsigned short*)(ws + 14 * MB);
    float*          Qf  = (float*)(ws + 16 * MB);           // 16 MiB each
    float*          Kf  = (float*)(ws + 32 * MB);
    float*          Vf  = (float*)(ws + 48 * MB);
    unsigned short* Qb  = (unsigned short*)(ws + 64 * MB);  // 8 MiB each
    unsigned short* Kb  = (unsigned short*)(ws + 72 * MB);
    unsigned short* Vt  = (unsigned short*)(ws + 80 * MB);
    unsigned short* Ob  = (unsigned short*)(ws + 88 * MB);  // 8 MiB

    const int M  = NB * SEQ;          // 4096
    const int nx = M * DMODEL;        // 4,194,304
    const int nw = DMODEL * DMODEL;   // 1,048,576

    k_f32_to_bf16<<<(nx + 255) / 256, 256, 0, stream>>>(x, xb, nx);
    k_f32_to_bf16_t<<<(nw + 255) / 256, 256, 0, stream>>>(Wq, wqt, DMODEL, DMODEL);
    k_f32_to_bf16_t<<<(nw + 255) / 256, 256, 0, stream>>>(Wk, wkt, DMODEL, DMODEL);
    k_f32_to_bf16_t<<<(nw + 255) / 256, 256, 0, stream>>>(Wv, wvt, DMODEL, DMODEL);
    k_f32_to_bf16_t<<<(nw + 255) / 256, 256, 0, stream>>>(Wo, wot, DMODEL, DMODEL);

    dim3 ggrid(DMODEL / 64, M / 256);   // (16, 16)
    k_gemm_bf16<<<ggrid, 256, 0, stream>>>(xb, wqt, Qf, M, DMODEL, DMODEL);
    k_gemm_bf16<<<ggrid, 256, 0, stream>>>(xb, wkt, Kf, M, DMODEL, DMODEL);
    k_gemm_bf16<<<ggrid, 256, 0, stream>>>(xb, wvt, Vf, M, DMODEL, DMODEL);

    k_rope_pack<<<(nx + 255) / 256, 256, 0, stream>>>(Qf, Kf, Vf, Qb, Kb, Vt);

    dim3 agrid(SEQ / 128, NB * NHEADS);  // (16, 32)
    k_attention<<<agrid, 256, 0, stream>>>(Qb, Kb, Vt, Ob);

    k_gemm_bf16<<<ggrid, 256, 0, stream>>>(Ob, wot, (float*)d_out, M, DMODEL, DMODEL);
}
